// GCN_36094905155901
// MI455X (gfx1250) — matrix operand order repaired and checked by reading, not compiled, not run on hardware
//
#include <hip/hip_runtime.h>
#include <hip/hip_bf16.h>

typedef __attribute__((ext_vector_type(16))) _Float16 v16h;
typedef __attribute__((ext_vector_type(8)))  _Float16 v8h;
typedef __attribute__((ext_vector_type(4)))  _Float16 v4h;
typedef __attribute__((ext_vector_type(8)))  float    v8f;

#define FDIM 128

// ---------------------------------------------------------------- utilities
__global__ void __launch_bounds__(256) zero_kernel(float* __restrict__ p, size_t n) {
    size_t i = (size_t)blockIdx.x * blockDim.x + threadIdx.x;
    if (i < n) p[i] = 0.0f;
}

__global__ void __launch_bounds__(256) deg_kernel(const long long* __restrict__ dst,
                                                  float* __restrict__ deg, int nedges) {
    int e = blockIdx.x * blockDim.x + threadIdx.x;
    if (e < nedges) atomicAdd(&deg[(int)dst[e]], 1.0f);
}

__global__ void __launch_bounds__(256) dinv_kernel(float* __restrict__ deg, int n) {
    int i = blockIdx.x * blockDim.x + threadIdx.x;
    if (i < n) deg[i] = rsqrtf(deg[i] + 1.0f);   // +1 self loop
}

// ---------------------------------------------------------------- WMMA GEMM
// XW[n, j] = sum_k H[n, k] * W[k, j];  H:[nrows,128] f32, W:[128,128] f32.
// Block = 128 threads (4 waves). Block computes 64 rows; each wave a 16x128 strip.
// W is staged TRANSPOSED in LDS so B fragments are contiguous K-runs (b128 loads).
__global__ void __launch_bounds__(128) gemm_xw(const float* __restrict__ H,
                                               const float* __restrict__ W,
                                               float* __restrict__ XW, int nrows) {
    __shared__ __align__(16) _Float16 sWt[FDIM][FDIM + 8];  // [j][k]  (transposed)
    __shared__ __align__(16) _Float16 sA [64  ][FDIM + 8];  // [m][k]

    const int tid  = threadIdx.x;
    const int wave = tid >> 5;
    const int lane = tid & 31;
    const int rowBlock = blockIdx.x * 64;

    // stage W transposed (f32 -> f16): thread t reads global row k=t, scatters column
    {
        const float4* wrow = (const float4*)(W + (size_t)tid * FDIM);
        #pragma unroll
        for (int j4 = 0; j4 < FDIM / 4; ++j4) {
            float4 w4 = wrow[j4];
            sWt[j4 * 4 + 0][tid] = (_Float16)w4.x;
            sWt[j4 * 4 + 1][tid] = (_Float16)w4.y;
            sWt[j4 * 4 + 2][tid] = (_Float16)w4.z;
            sWt[j4 * 4 + 3][tid] = (_Float16)w4.w;
        }
    }
    // stage A tile: thread t loads half-row (t/2, t%2), packed ds_store_b64
    {
        const int r    = tid >> 1;
        const int half = tid & 1;
        const int grow = rowBlock + r;
        const float* arow = H + (size_t)grow * FDIM + half * 64;
        #pragma unroll
        for (int j = 0; j < 64; j += 4) {
            float4 v = (grow < nrows) ? *(const float4*)(arow + j)
                                      : make_float4(0.f, 0.f, 0.f, 0.f);
            v4h p;
            p[0] = (_Float16)v.x; p[1] = (_Float16)v.y;
            p[2] = (_Float16)v.z; p[3] = (_Float16)v.w;
            *(v4h*)&sA[r][half * 64 + j] = p;
        }
    }
    __syncthreads();

    // per-lane fragment geometry (ISA 7.12.2, wave32)
    const int arow0 = wave * 16;
    const int mrow  = lane & 15;
    const int aKsel = (lane < 16) ? 0 : 8;    // A: 16x32 f16 -> halves i<8: K=aKsel+i; i>=8: K=aKsel+16+(i-8)
    const int bKsel = (lane < 16) ? 0 : 8;    // B: 32x16 f16 -> halves i<8: K=bKsel+i; i>=8: K=bKsel+16+(i-8)
    const int ncol  = lane & 15;

    v8f acc[8] = {};                          // 8 tiles x 8 VGPRs = 16x128 f32 strip

    #pragma unroll
    for (int kb = 0; kb < FDIM; kb += 32) {
        // A fragment: two contiguous 16B chunks -> 2x ds_load_b128
        const v8h alo = *(const v8h*)&sA[arow0 + mrow][kb + aKsel];
        const v8h ahi = *(const v8h*)&sA[arow0 + mrow][kb + aKsel + 16];
        const v16h afrag = __builtin_shufflevector(alo, ahi,
            0, 1, 2, 3, 4, 5, 6, 7, 8, 9, 10, 11, 12, 13, 14, 15);
        #pragma unroll
        for (int t = 0; t < 8; ++t) {
            // B fragment: two 8-half runs (at k and k+16) of a W^T row -> 2x ds_load_b128
            const v8h blo = *(const v8h*)&sWt[t * 16 + ncol][kb + bKsel];
            const v8h bhi = *(const v8h*)&sWt[t * 16 + ncol][kb + bKsel + 16];
            const v16h bfrag = __builtin_shufflevector(blo, bhi,
                0, 1, 2, 3, 4, 5, 6, 7, 8, 9, 10, 11, 12, 13, 14, 15);
            acc[t] = __builtin_amdgcn_wmma_f32_16x16x32_f16(
                false, afrag, false, bfrag, (short)0, acc[t], false, false);
        }
    }

    // store D: row = arow0 + (lane<16?0:8) + v, col = t*16 + (lane&15)
    const int droff = (lane < 16) ? 0 : 8;
    #pragma unroll
    for (int t = 0; t < 8; ++t) {
        #pragma unroll
        for (int v = 0; v < 8; ++v) {
            const int gr = rowBlock + arow0 + droff + v;
            if (gr < nrows)
                XW[(size_t)gr * FDIM + t * 16 + (lane & 15)] = acc[t][v];
        }
    }
}

// ---------------------------------------------------------------- aggregation
__global__ void __launch_bounds__(256) init_agg_kernel(float* __restrict__ agg,
                                                       const float* __restrict__ xw,
                                                       const float* __restrict__ dinv,
                                                       size_t total) {
    size_t i = (size_t)blockIdx.x * blockDim.x + threadIdx.x;
    if (i >= total) return;
    const int n = (int)(i >> 7);
    const float di = dinv[n];
    agg[i] = xw[i] * di * di;            // self-loop contribution
}

// one wave per edge: 32 lanes x float4 = full 128-float feature row
__global__ void __launch_bounds__(256) edge_scatter(const float* __restrict__ xw,
                                                    const float* __restrict__ dinv,
                                                    const long long* __restrict__ src,
                                                    const long long* __restrict__ dst,
                                                    float* __restrict__ agg, int nedges) {
    const int lane = threadIdx.x & 31;
    const long long e = (long long)blockIdx.x * 8 + (threadIdx.x >> 5);
    if (e >= nedges) return;
    const int s = (int)src[e];
    const int d = (int)dst[e];
    const float nrm = dinv[s] * dinv[d];
    // nudge next edge's source row toward L2 (global_prefetch_b8)
    if (e + 8 < nedges)
        __builtin_prefetch(xw + (size_t)((int)src[e + 8]) * FDIM + lane * 4, 0, 1);
    const float4 v = *(const float4*)(xw + (size_t)s * FDIM + lane * 4);
    float* ap = agg + (size_t)d * FDIM + lane * 4;
    atomicAdd(ap + 0, v.x * nrm);
    atomicAdd(ap + 1, v.y * nrm);
    atomicAdd(ap + 2, v.z * nrm);
    atomicAdd(ap + 3, v.w * nrm);
}

__global__ void __launch_bounds__(256) finalize_kernel(float* __restrict__ h,
                                                       const float* __restrict__ bias,
                                                       const float* __restrict__ gamma,
                                                       const float* __restrict__ beta,
                                                       const float* __restrict__ mean,
                                                       const float* __restrict__ var,
                                                       int apply_bn, size_t total) {
    size_t i = (size_t)blockIdx.x * blockDim.x + threadIdx.x;
    if (i >= total) return;
    const int f = (int)(i & (FDIM - 1));
    float v = h[i] + bias[f];
    if (apply_bn) {
        v = (v - mean[f]) * (gamma[f] * rsqrtf(var[f] + 1e-5f)) + beta[f];
        v = v > 0.0f ? v : 0.0f;
    }
    h[i] = v;
}

// ---------------------------------------------------------------- pooling + head
__global__ void __launch_bounds__(256) pool_kernel(const float* __restrict__ h,
                                                   const long long* __restrict__ batch,
                                                   float* __restrict__ sums,
                                                   float* __restrict__ counts, int nnodes) {
    const int lane = threadIdx.x & 31;
    const long long n = (long long)blockIdx.x * 8 + (threadIdx.x >> 5);
    if (n >= nnodes) return;
    const int b = (int)batch[n];
    const float4 v = *(const float4*)(h + (size_t)n * FDIM + lane * 4);
    float* sp = sums + (size_t)b * FDIM + lane * 4;
    atomicAdd(sp + 0, v.x);
    atomicAdd(sp + 1, v.y);
    atomicAdd(sp + 2, v.z);
    atomicAdd(sp + 3, v.w);
    if (lane == 0) atomicAdd(&counts[b], 1.0f);
}

__global__ void __launch_bounds__(256) head_kernel(const float* __restrict__ sums,
                                                   const float* __restrict__ counts,
                                                   const float* __restrict__ lin_w,
                                                   const float* __restrict__ lin_b,
                                                   float* __restrict__ out,
                                                   int ngraph, int nclass) {
    const int idx = blockIdx.x * blockDim.x + threadIdx.x;
    if (idx >= ngraph * nclass) return;
    const int g = idx / nclass, c = idx % nclass;
    float cnt = counts[g];
    cnt = cnt > 1.0f ? cnt : 1.0f;
    float a = 0.0f;
    const float* srow = sums + (size_t)g * FDIM;
    #pragma unroll 4
    for (int f = 0; f < FDIM; ++f) a += srow[f] * lin_w[f * nclass + c];
    out[idx] = a / cnt + lin_b[c];
}

// ---------------------------------------------------------------- launcher
extern "C" void kernel_launch(void* const* d_in, const int* in_sizes, int n_in,
                              void* d_out, int out_size, void* d_ws, size_t ws_size,
                              hipStream_t stream) {
    const float*      x      = (const float*)d_in[0];
    const long long*  ei     = (const long long*)d_in[1];
    const long long*  batch  = (const long long*)d_in[2];
    const float*      conv_w = (const float*)d_in[3];
    const float*      conv_b = (const float*)d_in[4];
    const float*      gamma  = (const float*)d_in[5];
    const float*      beta   = (const float*)d_in[6];
    const float*      mean   = (const float*)d_in[7];
    const float*      var    = (const float*)d_in[8];
    const float*      lin_w  = (const float*)d_in[9];
    const float*      lin_b  = (const float*)d_in[10];
    float*            out    = (float*)d_out;

    const int N = in_sizes[0] / FDIM;
    const int E = in_sizes[1] / 2;
    const int G = 512;                                   // graphs
    const int C = out_size / G;                          // classes
    const int L = 3;

    const long long* src = ei;
    const long long* dst = ei + E;

    // workspace layout (floats, everything 16B aligned)
    float* wsf    = (float*)d_ws;
    float* deg    = wsf;                                  // N   (becomes dinv)
    float* counts = deg + N;                              // G
    float* sums   = counts + G;                           // G*FDIM
    float* bufA   = sums + (size_t)G * FDIM;              // N*FDIM  (xw)
    float* bufB   = bufA + (size_t)N * FDIM;              // N*FDIM  (agg / h)

    const size_t total = (size_t)N * FDIM;

    // degrees -> D^{-1/2}
    zero_kernel<<<(N + 255) / 256, 256, 0, stream>>>(deg, (size_t)N);
    deg_kernel<<<(E + 255) / 256, 256, 0, stream>>>(dst, deg, E);
    dinv_kernel<<<(N + 255) / 256, 256, 0, stream>>>(deg, N);

    const float* hin = x;
    for (int l = 0; l < L; ++l) {
        gemm_xw<<<(N + 63) / 64, 128, 0, stream>>>(hin, conv_w + (size_t)l * FDIM * FDIM,
                                                   bufA, N);
        init_agg_kernel<<<(unsigned)((total + 255) / 256), 256, 0, stream>>>(bufB, bufA, deg, total);
        edge_scatter<<<(E + 7) / 8, 256, 0, stream>>>(bufA, deg, src, dst, bufB, E);
        finalize_kernel<<<(unsigned)((total + 255) / 256), 256, 0, stream>>>(
            bufB, conv_b + (size_t)l * FDIM,
            gamma + (size_t)l * FDIM, beta + (size_t)l * FDIM,
            mean + (size_t)l * FDIM, var + (size_t)l * FDIM,
            (l < L - 1) ? 1 : 0, total);
        hin = bufB;
    }

    // global mean pool + linear head
    zero_kernel<<<(unsigned)(((size_t)G * FDIM + 255) / 256), 256, 0, stream>>>(sums, (size_t)G * FDIM);
    zero_kernel<<<(G + 255) / 256, 256, 0, stream>>>(counts, (size_t)G);
    pool_kernel<<<(N + 7) / 8, 256, 0, stream>>>(bufB, batch, sums, counts, N);
    head_kernel<<<(G * C + 255) / 256, 256, 0, stream>>>(sums, counts, lin_w, lin_b, out, G, C);
}
